// SparsePeriodKernel_72215580115385
// MI455X (gfx1250) — compile-verified, matrix operand order checked by reading, not run on hardware
//
#include <hip/hip_runtime.h>
#include <hip/hip_bf16.h>
#include <stdint.h>

// ---------------------------------------------------------------------------
// SparsePeriod fused kernel for gfx1250 (MI455X):
//   depthwise conv1d(K=33, same) + residual  ->  per-phase 128x128 linear
// One block = one (batch, 16-channel tile). bf16 WMMA, fp32 accumulate.
// Stage 2: dual-phase per wave; A fragments preloaded per m-tile (graduated
// dscnt waits); stores use immediate offsets from one base pointer per m-tile.
// ---------------------------------------------------------------------------

typedef __attribute__((ext_vector_type(16))) __bf16 v16bf;
typedef __attribute__((ext_vector_type(8)))  __bf16 v8bf;
typedef __attribute__((ext_vector_type(8)))  float  v8f;

#define BATCH   32
#define LSEQ    4096
#define DMODEL  512
#define KW      33
#define NPER    128     // n (input phase length)
#define MPER    128     // m (output phase length)
#define PERIOD  32      // w (number of phases)
#define DTILE   16      // channels per block
#define XSTR    136     // bf16 row stride for X in LDS (banks 4i, 16B aligned)
#define WSTR    136     // bf16 row stride for W in LDS
#define RSTR    17      // f32 row stride for raw conv staging (conflict-free)
#define CHUNK   512     // l-values per conv chunk (= 16 n-columns)
#define NCHUNK  (LSEQ / CHUNK)
#define RAWROWS (CHUNK + 32)   // + K-1 halo

// LDS partition (bytes)
#define XL_OFF   0
#define XL_ELEMS (PERIOD * DTILE * XSTR)           // 69632 u16 -> 139264 B
#define WL_OFF   (XL_OFF + XL_ELEMS * 2)
#define WL_ELEMS (MPER * WSTR)                     // 17408 u16 -> 34816 B
#define RAW_OFF  (WL_OFF + WL_ELEMS * 2)
#define RAW_ELEMS (RAWROWS * RSTR)                 // 9248 f32 -> 36992 B
#define CW_OFF   (RAW_OFF + RAW_ELEMS * 4)
#define CW_ELEMS (DTILE * 34)                      // 544 f32 -> 2176 B
#define SMEM_BYTES (CW_OFF + CW_ELEMS * 4)         // 213248 B (< 320 KB/WGP)

__device__ __forceinline__ unsigned short f2bf(float f) {
  unsigned int u = __float_as_uint(f);
  u += 0x7FFFu + ((u >> 16) & 1u);   // round-to-nearest-even
  return (unsigned short)(u >> 16);
}

// One lane's 16 bf16 values of a 16x32 WMMA operand fragment from an LDS row.
// CDNA5 16-bit layout: lanes 0-15 hold K = kbase+{0..7} and kbase+{16..23};
// lanes 16-31 the +8-shifted ranges (ksel folds the +8 into kbase).
// Two 16B ds loads per lane; padded row stride -> all 32 lanes distinct banks.
__device__ __forceinline__ v16bf load_frag(const unsigned short* rowp, int kbase) {
  v8bf lo = *(const v8bf*)(rowp + kbase);
  v8bf hi = *(const v8bf*)(rowp + kbase + 16);
  return __builtin_shufflevector(lo, hi,
      0, 1, 2, 3, 4, 5, 6, 7, 8, 9, 10, 11, 12, 13, 14, 15);
}

__global__ void sparse_period_fused(const float* __restrict__ h,
                                    const float* __restrict__ conv_w,
                                    const float* __restrict__ bb_w,
                                    float* __restrict__ out) {
  extern __shared__ unsigned char smem[];
  unsigned short* Xl  = (unsigned short*)(smem + XL_OFF);   // X[w][d][n] bf16
  unsigned short* Wl  = (unsigned short*)(smem + WL_OFF);   // W[m][n]   bf16
  float*          raw = (float*)(smem + RAW_OFF);           // conv staging f32
  float*          cw  = (float*)(smem + CW_OFF);            // conv weights f32

  const int tid = threadIdx.x;
  const int d0  = blockIdx.x * DTILE;
  const int bb  = blockIdx.y;

  // ---- one-time loads: backbone W (->bf16) and this tile's conv weights ----
  for (int idx = tid; idx < MPER * NPER; idx += 256) {
    int m = idx >> 7, n = idx & 127;
    Wl[m * WSTR + n] = f2bf(bb_w[m * NPER + n]);
  }
  for (int idx = tid; idx < DTILE * KW; idx += 256) {
    int dd = idx / KW, k = idx - dd * KW;
    cw[dd * 34 + k] = conv_w[(d0 + dd) * KW + k];
  }

  const int dd = tid & 15;   // channel within tile
  const int lg = tid >> 4;   // l-group (16 groups)
  __syncthreads();

  // register-cache the 33 conv taps for this lane's channel (broadcast pairs)
  float wreg[KW];
#pragma unroll
  for (int k = 0; k < KW; ++k) wreg[k] = cw[dd * 34 + k];

  // ---- stage 1: depthwise conv + residual, streamed in 512-l chunks ----
  for (int c = 0; c < NCHUNK; ++c) {
    const int lbase = c * CHUNK - 16;
    // stage raw h chunk (+halo) into LDS, zero-padded at sequence edges
    for (int idx = tid; idx < RAWROWS * DTILE; idx += 256) {
      int r = idx >> 4, d2 = idx & 15;
      int l = lbase + r;
      float v = 0.0f;
      if (l >= 0 && l < LSEQ)
        v = h[((size_t)bb * LSEQ + l) * DMODEL + d0 + d2];
      raw[r * RSTR + d2] = v;
    }
    if (c + 1 < NCHUNK)   // hint next chunk into cache (global_prefetch_b8)
      __builtin_prefetch(&h[((size_t)bb * LSEQ + (c + 1) * CHUNK + lg * 32) * DMODEL + d0 + dd], 0, 0);
    __syncthreads();

#pragma unroll 4
    for (int i = 0; i < CHUNK / 16; ++i) {
      int ll = lg + i * 16;                          // 0..511 within chunk
      float acc = raw[(ll + 16) * RSTR + dd];        // residual term
#pragma unroll
      for (int k = 0; k < KW; ++k)
        acc += wreg[k] * raw[(ll + k) * RSTR + dd];
      int l = c * CHUNK + ll;
      int w = l & 31, n = l >> 5;                    // phase / phase-index
      Xl[(w * DTILE + dd) * XSTR + n] = f2bf(acc);
    }
    __syncthreads();
  }

  // ---- stage 2: per-phase GEMM  Y^T[m,d] = W[m,:] x X[d,:]^T  via WMMA ----
  // A = W tile (16m x 32n), B = X^T tile (32n x 16d), C = 16m x 16d f32.
  // Each wave owns phases {wave, wave+8, wave+16, wave+24}, processed as two
  // pairs so every A fragment feeds two WMMAs. m-loop deliberately NOT
  // unrolled (no spill-hoisting); all 4 A fragments preloaded per m-tile so
  // the 8 ds loads issue ahead of the WMMA chain (graduated dscnt waits).
  const int lane = tid & 31;
  const int wave = tid >> 5;
  const int lrow = lane & 15;            // A: m row / B,C: d col
  const int ksel = (lane >> 4) << 3;     // +8 K-offset for upper lane half

#pragma unroll 1
  for (int wp = 0; wp < 2; ++wp) {
    const int wt0 = wave + wp * 16;      // this pass: phases wt0 and wt0+8
    const unsigned short* xr0 = Xl + ((wt0)     * DTILE + lrow) * XSTR;
    const unsigned short* xr1 = Xl + ((wt0 + 8) * DTILE + lrow) * XSTR;
    v16bf b0[4], b1[4];
#pragma unroll
    for (int kc = 0; kc < 4; ++kc) {
      b0[kc] = load_frag(xr0, kc * 32 + ksel);
      b1[kc] = load_frag(xr1, kc * 32 + ksel);
    }

    // per-wp output base: lane column (lrow) + lane-half m offset + phase wt0
    float* const obase = out + ((size_t)bb * LSEQ +
                                (size_t)((lane >> 4) << 3) * PERIOD + wt0) * DMODEL +
                         d0 + lrow;

#pragma unroll 1
    for (int mt = 0; mt < 8; ++mt) {
      const unsigned short* wr = Wl + (mt * 16 + lrow) * WSTR;
      v16bf a[4];
#pragma unroll
      for (int kc = 0; kc < 4; ++kc)
        a[kc] = load_frag(wr, kc * 32 + ksel);

      v8f c0 = {};
      v8f c1 = {};
#pragma unroll
      for (int kc = 0; kc < 4; ++kc) {
        c0 = __builtin_amdgcn_wmma_f32_16x16x32_bf16(
            false, a[kc], false, b0[kc], (short)0, c0, false, false);
        c1 = __builtin_amdgcn_wmma_f32_16x16x32_bf16(
            false, a[kc], false, b1[kc], (short)0, c1, false, false);
      }

      // C layout: VGPR r -> (m = mt*16 + r + 8*(lane>=16), d = d0 + lane%16).
      // Single base pointer per m-tile; all 16 stores are immediate offsets
      // (r * 64KiB, +16KiB for phase wt0+8) -> no per-store address VALU.
      float* const op = obase + (size_t)mt * (16 * PERIOD * DMODEL);
#pragma unroll
      for (int r = 0; r < 8; ++r) {
        op[(size_t)(r * PERIOD * DMODEL)]               = c0[r];
        op[(size_t)(r * PERIOD * DMODEL + 8 * DMODEL)]  = c1[r];
      }
    }
  }
}

extern "C" void kernel_launch(void* const* d_in, const int* in_sizes, int n_in,
                              void* d_out, int out_size, void* d_ws, size_t ws_size,
                              hipStream_t stream) {
  (void)in_sizes; (void)n_in; (void)d_ws; (void)ws_size; (void)out_size;
  const float* h   = (const float*)d_in[0];
  const float* cwp = (const float*)d_in[1];
  const float* bwp = (const float*)d_in[2];
  float* out = (float*)d_out;

  hipFuncSetAttribute(reinterpret_cast<const void*>(sparse_period_fused),
                      hipFuncAttributeMaxDynamicSharedMemorySize, SMEM_BYTES);

  dim3 grid(DMODEL / DTILE, BATCH);   // 32 x 32 = 1024 blocks
  sparse_period_fused<<<grid, dim3(256), SMEM_BYTES, stream>>>(h, cwp, bwp, out);
}
